// GPT_28664611734078
// MI455X (gfx1250) — compile-verified, matrix operand order checked by reading, not run on hardware
//
#include <hip/hip_runtime.h>
#include <math.h>

// GPT-2-ish forward for MI455X (gfx1250, wave32, WMMA bf16, TDM-staged GEMM)
// V=32000 T=1024 B=2 L=6 C=768 H=12 DFF=3072 DH=64

#define VV   32000
#define TT   1024
#define BB   2
#define LLY  6
#define CCH  768
#define HH   12
#define DFFD 3072
#define DHD  64

typedef __attribute__((ext_vector_type(16))) __bf16          v16bf;
typedef __attribute__((ext_vector_type(8)))  float           v8f;
typedef __attribute__((ext_vector_type(8)))  unsigned short  v8us;
typedef __attribute__((ext_vector_type(4)))  unsigned int    v4u;
typedef __attribute__((ext_vector_type(8)))  int             v8i;
typedef __attribute__((ext_vector_type(4)))  int             v4i;
typedef unsigned short u16;

union FragU { v16bf v; v8us h[2]; };

static __device__ __forceinline__ u16 f2bf(float f) {
  union { float f; unsigned u; } u; u.f = f;
  unsigned r = u.u + 0x7FFFu + ((u.u >> 16) & 1u);   // round-to-nearest-even
  return (u16)(r >> 16);
}

// 16x32 bf16 fragment (A-layout; also used as B operand for X @ W^T since W is [N,K] row-major):
// lane m (0-15) holds K = k0..k0+7 and k0+16..k0+23 of row row0+m;
// lane m+16 holds K = k0+8..k0+15 and k0+24..k0+31 of the same row.
static __device__ __forceinline__ v16bf load_frag(const u16* base, int ld, int row0, int k0) {
  int lane = threadIdx.x & 31;
  int r    = row0 + (lane & 15);
  int kb   = k0 + ((lane >> 4) << 3);
  const u16* p = base + (size_t)r * ld + kb;
  FragU f;
  f.h[0] = *(const v8us*)(p);
  f.h[1] = *(const v8us*)(p + 16);
  return f.v;
}

static __device__ __forceinline__ v8f wmma_bf(v16bf a, v16bf b, v8f c) {
  return __builtin_amdgcn_wmma_f32_16x16x32_bf16(false, a, false, b, (short)0, c, false, false);
}

// ---------------------------------------------------------------- TDM: 128x32 bf16 tile, row stride ld (elems),
// DMA'd global->LDS with 16B pad per 64B row => LDS row stride = 40 elems (bank-conflict-free fragments).
static __device__ __forceinline__ void tdm_load_tile(const u16* gptr, int ldElems, unsigned ldsByteOff) {
  unsigned long long ga = (unsigned long long)(const void*)gptr;
  // group0: count=1 | lds_addr | global_addr[56:0] | type=2
  v4u g0 = { 1u,
             ldsByteOff,
             (unsigned)ga,
             ((unsigned)((ga >> 32) & 0x01FFFFFFull)) | 0x80000000u };
  // group1: data_size=1(2B), pad_enable, pad_interval=3(16 DW), pad_amount=3(4 DW),
  //         tensor_dim0/1 = huge (tiles always in-bounds), tile_dim0=32, tile_dim1=128,
  //         tensor_dim0_stride = ldElems
  const unsigned long long td = 1ull << 30;
  unsigned long long st0 = (unsigned long long)ldElems;
  unsigned d0 = (1u << 16) | (1u << 20) | (3u << 22) | (3u << 25);
  unsigned d1 = (unsigned)((td & 0xFFFFull) << 16);                    // tensor_dim0[15:0] @63:48
  unsigned d2 = (unsigned)((td >> 16) & 0xFFFFull)                     // tensor_dim0[31:16]
              | (unsigned)((td & 0xFFFFull) << 16);                    // tensor_dim1[15:0]
  unsigned d3 = (unsigned)((td >> 16) & 0xFFFFull) | (32u << 16);      // tensor_dim1[31:16] | tile_dim0
  unsigned d4 = 128u;                                                  // tile_dim1 (tile_dim2=0)
  unsigned d5 = (unsigned)(st0 & 0xFFFFFFFFull);                       // dim0_stride[31:0]
  unsigned d6 = (unsigned)((st0 >> 32) & 0xFFFFull);                   // dim0_stride[47:32]
  v8i g1 = { (int)d0, (int)d1, (int)d2, (int)d3, (int)d4, (int)d5, (int)d6, 0 };
  // group2: tensor_dim2/3 huge, stride2=0, tile_dim3=0
  v4i g2 = { (int)0x40000000, (int)0x40000000, 0, 0 };
  // group3: stride3=0, tensor_dim4 huge (bits 79:48), tile_dim4=0
  v4i g3 = { 0, 0, (int)0x4000, 0 };
  // trailing descriptor group (unused for a 2D tile) — zero
  v8i g4 = { 0, 0, 0, 0, 0, 0, 0, 0 };
  __builtin_amdgcn_tensor_load_to_lds(g0, g1, g2, g3, g4, 0);
}

// ---------------------------------------------------------------- elementwise
__global__ __launch_bounds__(256) void cvt_f32_bf16(const float* __restrict__ in,
                                                    u16* __restrict__ out, size_t n) {
  for (size_t i = blockIdx.x * 256ull + threadIdx.x; i < n; i += (size_t)gridDim.x * 256ull)
    out[i] = f2bf(in[i]);
}

__global__ __launch_bounds__(256) void embed_k(const int* __restrict__ idx,
                                               const float* __restrict__ tok,
                                               const float* __restrict__ pos,
                                               float* __restrict__ x) {
  size_t i = blockIdx.x * 256ull + threadIdx.x;
  if (i >= (size_t)BB * TT * CCH) return;
  int c = (int)(i % CCH);
  int t = (int)((i / CCH) % TT);
  size_t bt = i / CCH;
  x[i] = tok[(size_t)idx[bt] * CCH + c] + pos[(size_t)t * CCH + c];
}

// ---------------------------------------------------------------- layernorm -> bf16
__global__ __launch_bounds__(256) void ln_bf16(const float* __restrict__ x,
                                               const float* __restrict__ g,
                                               const float* __restrict__ b,
                                               u16* __restrict__ out, int CN) {
  int row = blockIdx.x;
  const float* xr = x + (size_t)row * CN;
  __shared__ float s1[256], s2[256];
  float a = 0.f, q = 0.f;
  for (int c = threadIdx.x; c < CN; c += 256) { float v = xr[c]; a += v; q += v * v; }
  s1[threadIdx.x] = a; s2[threadIdx.x] = q; __syncthreads();
  for (int off = 128; off > 0; off >>= 1) {
    if ((int)threadIdx.x < off) { s1[threadIdx.x] += s1[threadIdx.x + off];
                                  s2[threadIdx.x] += s2[threadIdx.x + off]; }
    __syncthreads();
  }
  float mu  = s1[0] / CN;
  float var = s2[0] / CN - mu * mu;
  float rs  = rsqrtf(var + 1e-5f);
  for (int c = threadIdx.x; c < CN; c += 256)
    out[(size_t)row * CN + c] = f2bf((xr[c] - mu) * rs * g[c] + b[c]);
}

// ---------------------------------------------------------------- WMMA GEMM: out = A[M,K] @ W[N,K]^T (+bias)(+epi)
// TDM double-buffered LDS staging of 128x32 tiles; 4 waves = 128x128 block tile.
// EPI: 0 = fp32 out + bias; 1 = GELU -> bf16 out; 2 = residual add fp32 (in-place ok); 3 = fp32 out, no bias
template <int EPI>
__global__ __launch_bounds__(128)
void gemm_wmma(const u16* __restrict__ A, int lda,
               const u16* __restrict__ W, int ldw,
               const float* __restrict__ bias,
               const float* __restrict__ res,
               float* __restrict__ outf,
               u16* __restrict__ outb,
               int M, int N, int K) {
  __shared__ u16 sA[2][128][40];   // 40 = 32 + 4DW pad inserted by TDM per 16DW row
  __shared__ u16 sB[2][128][40];
  int lane = threadIdx.x & 31;
  int wave = threadIdx.x >> 5;
  int mBlk = blockIdx.y * 128;
  int nBlk = blockIdx.x * 128;
  int m0 = (wave >> 1) * 64;       // block-local
  int n0 = (wave & 1) * 64;
  (void)M;

  v8f acc[4][4];
#pragma unroll
  for (int i = 0; i < 4; ++i)
#pragma unroll
    for (int j = 0; j < 4; ++j)
#pragma unroll
      for (int r = 0; r < 8; ++r) acc[i][j][r] = 0.f;

  const bool issuer = (wave == 0);
  unsigned offA0 = (unsigned)(size_t)&sA[0][0][0];
  unsigned offA1 = (unsigned)(size_t)&sA[1][0][0];
  unsigned offB0 = (unsigned)(size_t)&sB[0][0][0];
  unsigned offB1 = (unsigned)(size_t)&sB[1][0][0];

  if (issuer) {
    tdm_load_tile(A + (size_t)mBlk * lda, lda, offA0);
    tdm_load_tile(W + (size_t)nBlk * ldw, ldw, offB0);
    __builtin_amdgcn_s_wait_tensorcnt(0);
  }
  __syncthreads();

  int buf = 0;
  for (int k0 = 0; k0 < K; k0 += 32) {
    if (issuer && (k0 + 32 < K)) {     // prefetch next K-slab into other buffer (overlaps WMMA)
      tdm_load_tile(A + (size_t)mBlk * lda + (k0 + 32), lda, buf ? offA0 : offA1);
      tdm_load_tile(W + (size_t)nBlk * ldw + (k0 + 32), ldw, buf ? offB0 : offB1);
    }
    const u16* aBase = &sA[buf][0][0];
    const u16* bBase = &sB[buf][0][0];
    v16bf a[4], b[4];
#pragma unroll
    for (int i = 0; i < 4; ++i) a[i] = load_frag(aBase, 40, m0 + 16 * i, 0);
#pragma unroll
    for (int j = 0; j < 4; ++j) b[j] = load_frag(bBase, 40, n0 + 16 * j, 0);
#pragma unroll
    for (int i = 0; i < 4; ++i)
#pragma unroll
      for (int j = 0; j < 4; ++j) acc[i][j] = wmma_bf(a[i], b[j], acc[i][j]);

    if (issuer) __builtin_amdgcn_s_wait_tensorcnt(0);
    __syncthreads();
    buf ^= 1;
  }

  int cb = lane & 15;
  int rb = (lane >> 4) << 3;
#pragma unroll
  for (int i = 0; i < 4; ++i)
#pragma unroll
    for (int j = 0; j < 4; ++j) {
      int n = nBlk + n0 + 16 * j + cb;
      float bv = (EPI == 3) ? 0.f : bias[n];
#pragma unroll
      for (int r = 0; r < 8; ++r) {
        int m = mBlk + m0 + 16 * i + rb + r;
        float v = acc[i][j][r] + bv;
        size_t o = (size_t)m * N + n;
        if (EPI == 0 || EPI == 3) {
          outf[o] = v;
        } else if (EPI == 1) {
          outb[o] = f2bf(0.5f * v * (1.f + erff(v * 0.70710678118f)));
        } else {
          outf[o] = res[o] + v;
        }
      }
    }
}

// ---------------------------------------------------------------- QKV rearrange (fp32 qkv -> bf16 Q,K,V^T per head)
__global__ __launch_bounds__(256) void qkv_split(const float* __restrict__ qkvf,
                                                 u16* __restrict__ Qb,
                                                 u16* __restrict__ Kb,
                                                 u16* __restrict__ Vt) {
  size_t i = blockIdx.x * 256ull + threadIdx.x;
  if (i >= (size_t)BB * TT * CCH) return;
  int c = (int)(i % CCH);
  int t = (int)((i / CCH) % TT);
  int b = (int)(i / ((size_t)CCH * TT));
  int h = c / DHD, d = c % DHD;
  const float* base = qkvf + ((size_t)b * TT + t) * (3 * CCH);
  size_t bh = (size_t)b * HH + h;
  Qb[(bh * TT + t) * DHD + d] = f2bf(base[c]);
  Kb[(bh * TT + t) * DHD + d] = f2bf(base[CCH + c]);
  Vt[(bh * DHD + d) * TT + t] = f2bf(base[2 * CCH + c]);
}

// ---------------------------------------------------------------- flash attention (causal), 1 wave = 16 query rows
__global__ __launch_bounds__(128)
void attn_flash(const u16* __restrict__ Qb, const u16* __restrict__ Kb,
                const u16* __restrict__ Vt, u16* __restrict__ attn_out) {
  __shared__ u16 pbuf[4][16][72];   // per-wave 16x64 P tile (bf16), padded
  int lane = threadIdx.x & 31;
  int wave = threadIdx.x >> 5;
  int bh = blockIdx.y;
  int b = bh / HH, h = bh % HH;
  int q0 = blockIdx.x * 64 + wave * 16;
  const u16* Qp = Qb + (size_t)bh * TT * DHD;
  const u16* Kp = Kb + (size_t)bh * TT * DHD;
  const u16* Vp = Vt + (size_t)bh * DHD * TT;

  v16bf qa0 = load_frag(Qp, DHD, q0, 0);
  v16bf qa1 = load_frag(Qp, DHD, q0, 32);

  v8f o[4];
#pragma unroll
  for (int df = 0; df < 4; ++df)
#pragma unroll
    for (int r = 0; r < 8; ++r) o[df][r] = 0.f;
  float mr[8], lr[8];
#pragma unroll
  for (int r = 0; r < 8; ++r) { mr[r] = -3.0e38f; lr[r] = 0.f; }

  int cb = lane & 15;
  int rb = (lane >> 4) << 3;
  int jmax = blockIdx.x;   // uniform across all 4 waves

  for (int j = 0; j <= jmax; ++j) {
    int s0 = j * 64;
    v8f S[4];
#pragma unroll
    for (int nf = 0; nf < 4; ++nf) {
      v8f s;
#pragma unroll
      for (int r = 0; r < 8; ++r) s[r] = 0.f;
      v16bf kf0 = load_frag(Kp, DHD, s0 + nf * 16, 0);
      v16bf kf1 = load_frag(Kp, DHD, s0 + nf * 16, 32);
      s = wmma_bf(qa0, kf0, s);
      s = wmma_bf(qa1, kf1, s);
      S[nf] = s;
    }
    const float scale = 0.125f;   // 1/sqrt(64)
    if (j == jmax) {
#pragma unroll
      for (int nf = 0; nf < 4; ++nf)
#pragma unroll
        for (int r = 0; r < 8; ++r) {
          int sc = s0 + nf * 16 + cb;
          int qm = q0 + rb + r;
          S[nf][r] = (sc <= qm) ? S[nf][r] * scale : -3.0e38f;
        }
    } else {
#pragma unroll
      for (int nf = 0; nf < 4; ++nf)
#pragma unroll
        for (int r = 0; r < 8; ++r) S[nf][r] *= scale;
    }

    // online softmax (per row r; 16 lanes of a half-group hold 16 cols each)
#pragma unroll
    for (int r = 0; r < 8; ++r) {
      float v = fmaxf(fmaxf(S[0][r], S[1][r]), fmaxf(S[2][r], S[3][r]));
#pragma unroll
      for (int off = 8; off >= 1; off >>= 1) v = fmaxf(v, __shfl_xor(v, off, 32));
      float mn = fmaxf(mr[r], v);
      float f = __expf(mr[r] - mn);
      mr[r] = mn;
      float rs = 0.f;
#pragma unroll
      for (int nf = 0; nf < 4; ++nf) {
        float p = __expf(S[nf][r] - mn);
        S[nf][r] = p;
        rs += p;
      }
#pragma unroll
      for (int off = 8; off >= 1; off >>= 1) rs += __shfl_xor(rs, off, 32);
      lr[r] = lr[r] * f + rs;
#pragma unroll
      for (int df = 0; df < 4; ++df) o[df][r] *= f;
    }

    // C/D-layout P -> LDS (row-major bf16) -> reload as A-layout fragments
#pragma unroll
    for (int nf = 0; nf < 4; ++nf)
#pragma unroll
      for (int r = 0; r < 8; ++r)
        pbuf[wave][rb + r][nf * 16 + cb] = f2bf(S[nf][r]);
    asm volatile("s_wait_dscnt 0x0" ::: "memory");

#pragma unroll
    for (int ks = 0; ks < 2; ++ks) {
      v16bf pa = load_frag(&pbuf[wave][0][0], 72, 0, ks * 32);
#pragma unroll
      for (int df = 0; df < 4; ++df) {
        v16bf vb = load_frag(Vp, TT, df * 16, s0 + ks * 32);   // V^T rows = output dims d
        o[df] = wmma_bf(pa, vb, o[df]);
      }
    }
  }

  // normalize + scatter to [B,T,C] bf16
#pragma unroll
  for (int df = 0; df < 4; ++df)
#pragma unroll
    for (int r = 0; r < 8; ++r) {
      int d = df * 16 + cb;
      int m = q0 + rb + r;
      attn_out[((size_t)b * TT + m) * CCH + h * DHD + d] = f2bf(o[df][r] / lr[r]);
    }
}

// ---------------------------------------------------------------- host
extern "C" void kernel_launch(void* const* d_in, const int* in_sizes, int n_in,
                              void* d_out, int out_size, void* d_ws, size_t ws_size,
                              hipStream_t stream) {
  (void)in_sizes; (void)n_in; (void)out_size; (void)ws_size;
  const int*   idx   = (const int*)  d_in[0];
  const float* tok   = (const float*)d_in[1];
  const float* pos   = (const float*)d_in[2];
  const float* ln1g  = (const float*)d_in[3];
  const float* ln1b  = (const float*)d_in[4];
  const float* qkvw  = (const float*)d_in[5];
  const float* qkvb  = (const float*)d_in[6];
  const float* projw = (const float*)d_in[7];
  const float* projb = (const float*)d_in[8];
  const float* ln2g  = (const float*)d_in[9];
  const float* ln2b  = (const float*)d_in[10];
  const float* ff1w  = (const float*)d_in[11];
  const float* ff1b  = (const float*)d_in[12];
  const float* ff2w  = (const float*)d_in[13];
  const float* ff2b  = (const float*)d_in[14];
  const float* lnfg  = (const float*)d_in[15];
  const float* lnfb  = (const float*)d_in[16];
  const float* headw = (const float*)d_in[17];

  char* ws = (char*)d_ws;
  size_t off = 0;
  auto alloc = [&](size_t bytes) -> void* {
    void* p = ws + off;
    off = (off + bytes + 255) & ~(size_t)255;
    return p;
  };

  const int M = BB * TT;                    // 2048
  u16* wq = (u16*)alloc((size_t)LLY * 3 * CCH * CCH * 2);
  u16* wp = (u16*)alloc((size_t)LLY * CCH * CCH * 2);
  u16* w1 = (u16*)alloc((size_t)LLY * DFFD * CCH * 2);
  u16* w2 = (u16*)alloc((size_t)LLY * CCH * DFFD * 2);
  u16* wh = (u16*)alloc((size_t)VV * CCH * 2);
  float* x    = (float*)alloc((size_t)M * CCH * 4);
  u16*   xb   = (u16*)  alloc((size_t)M * CCH * 2);
  u16*   hb   = (u16*)  alloc((size_t)M * DFFD * 2);
  float* qkvf = (float*)alloc((size_t)M * 3 * CCH * 4);
  u16*   Qbuf = (u16*)  alloc((size_t)BB * HH * TT * DHD * 2);
  u16*   Kbuf = (u16*)  alloc((size_t)BB * HH * TT * DHD * 2);
  u16*   Vtb  = (u16*)  alloc((size_t)BB * HH * DHD * TT * 2);
  u16*   atb  = (u16*)  alloc((size_t)M * CCH * 2);

  // weight conversion (fp32 -> bf16), every call (deterministic)
  cvt_f32_bf16<<<4096, 256, 0, stream>>>(qkvw, wq, (size_t)LLY * 3 * CCH * CCH);
  cvt_f32_bf16<<<4096, 256, 0, stream>>>(projw, wp, (size_t)LLY * CCH * CCH);
  cvt_f32_bf16<<<4096, 256, 0, stream>>>(ff1w, w1, (size_t)LLY * DFFD * CCH);
  cvt_f32_bf16<<<4096, 256, 0, stream>>>(ff2w, w2, (size_t)LLY * CCH * DFFD);
  cvt_f32_bf16<<<8192, 256, 0, stream>>>(headw, wh, (size_t)VV * CCH);

  // embedding
  {
    size_t n = (size_t)M * CCH;
    embed_k<<<(unsigned)((n + 255) / 256), 256, 0, stream>>>(idx, tok, pos, x);
  }

  dim3 blk(128);
  for (int l = 0; l < LLY; ++l) {
    // ln1 -> xb
    ln_bf16<<<M, 256, 0, stream>>>(x, ln1g + (size_t)l * CCH, ln1b + (size_t)l * CCH, xb, CCH);
    // qkv = xb @ qkv_w^T + b  (fp32)
    gemm_wmma<0><<<dim3(3 * CCH / 128, M / 128), blk, 0, stream>>>(
        xb, CCH, wq + (size_t)l * 3 * CCH * CCH, CCH,
        qkvb + (size_t)l * 3 * CCH, nullptr, qkvf, nullptr, M, 3 * CCH, CCH);
    // split / transpose heads
    {
      size_t n = (size_t)M * CCH;
      qkv_split<<<(unsigned)((n + 255) / 256), 256, 0, stream>>>(qkvf, Qbuf, Kbuf, Vtb);
    }
    // attention
    attn_flash<<<dim3(TT / 64, BB * HH), blk, 0, stream>>>(Qbuf, Kbuf, Vtb, atb);
    // x += attn @ proj_w^T + b
    gemm_wmma<2><<<dim3(CCH / 128, M / 128), blk, 0, stream>>>(
        atb, CCH, wp + (size_t)l * CCH * CCH, CCH,
        projb + (size_t)l * CCH, x, x, nullptr, M, CCH, CCH);
    // ln2 -> xb
    ln_bf16<<<M, 256, 0, stream>>>(x, ln2g + (size_t)l * CCH, ln2b + (size_t)l * CCH, xb, CCH);
    // hb = gelu(xb @ ff1^T + b) (bf16)
    gemm_wmma<1><<<dim3(DFFD / 128, M / 128), blk, 0, stream>>>(
        xb, CCH, w1 + (size_t)l * DFFD * CCH, CCH,
        ff1b + (size_t)l * DFFD, nullptr, nullptr, hb, M, DFFD, CCH);
    // x += hb @ ff2^T + b
    gemm_wmma<2><<<dim3(CCH / 128, M / 128), blk, 0, stream>>>(
        hb, DFFD, w2 + (size_t)l * CCH * DFFD, DFFD,
        ff2b + (size_t)l * CCH, x, x, nullptr, M, CCH, DFFD);
  }

  // final LN + head
  ln_bf16<<<M, 256, 0, stream>>>(x, lnfg, lnfb, xb, CCH);
  gemm_wmma<3><<<dim3(VV / 128, M / 128), blk, 0, stream>>>(
      xb, CCH, wh, CCH, nullptr, nullptr, (float*)d_out, nullptr, M, VV, CCH);
}
